// RegionLearner_76252849373237
// MI455X (gfx1250) — matrix operand order repaired and checked by reading, not compiled, not run on hardware
//
#include <hip/hip_runtime.h>
#include <hip/hip_bf16.h>

// ---------------------------------------------------------------------------
// RegionLearner for MI455X (gfx1250, wave32, WMMA + TDM + async-LDS).
//
//   region_mask[b,s,h,w] = conv3x3(x, w)[b,s,h,w] + bias[s]          (f32 out)
//   learned[b,c,s]       = (1/784) * sum_hw x[b,c,hw]*mask[b,s,hw]   (f32 out)
//
// Both GEMMs run on v_wmma_f32_16x16x32_bf16 with a bf16 hi/lo split
// (3 WMMA terms: hh + hl + lh) for ~fp32 accuracy at bf16-roofline speed.
//
// Conv workgroup = 7 waves (one image's 49 pixel-tiles split 7x7). Packed
// A-fragment weights for the current 3x3 tap are staged into LDS once per
// block: hi tile via one TENSOR_LOAD_TO_LDS (TDM, TENSORcnt), lo tile via
// per-lane global_load_async_to_lds_b128 (ASYNCcnt). All 7 waves then read
// A-fragments from LDS (ds_load_b128) -- 7x less L2 traffic on the A side.
//
// d_ws layout (bf16 elements):
//   wph [216*32*16]  pre-packed A-fragment weights (hi)
//   wpl [216*32*16]  pre-packed A-fragment weights (lo)
//   mh  [64*8*800]   region mask bf16 hi, K padded 784 -> 800
//   ml  [64*8*800]   region mask bf16 lo
// Total ~2.1 MB.
// ---------------------------------------------------------------------------

typedef __attribute__((ext_vector_type(16))) __bf16 v16bf;
typedef __attribute__((ext_vector_type(8)))  __bf16 v8bf;
typedef __attribute__((ext_vector_type(8)))  float  v8f;
typedef __attribute__((ext_vector_type(4)))  float  v4f;
typedef __attribute__((ext_vector_type(4)))  unsigned v4u;
typedef __attribute__((ext_vector_type(8)))  unsigned v8u;

#define NB   64
#define NC   768
#define NS   8
#define NHW  784      // 28*28
#define NK   6912     // NC*9
#define NCH  216      // NK/32 K-chunks for conv
#define MPAD 800      // NHW padded to multiple of 32
#define WPK  (NCH*32*16)   // packed-weight elements per (hi|lo)
#define GRP  (24*32*16)    // packed elements per 3x3 tap group (12288)

__device__ __forceinline__ __bf16 bhi(float v) { return (__bf16)v; }
__device__ __forceinline__ __bf16 blo(float v, __bf16 hi) { return (__bf16)(v - (float)hi); }

// gfx1250 async copy: 16B global -> LDS, per-lane addresses, tracked by ASYNCcnt.
__device__ __forceinline__ void async_copy16(const void* g, unsigned lds_off) {
    asm volatile("global_load_async_to_lds_b128 %0, %1, off"
                 :: "v"(lds_off), "v"((unsigned long long)(size_t)g)
                 : "memory");
}
__device__ __forceinline__ void wait_asynccnt0() {
    asm volatile("s_wait_asynccnt 0x0" ::: "memory");
}

// gfx1250 Tensor Data Mover: DMA a contiguous run of `ndw` dwords from global
// memory into LDS with one instruction. 1-D D# (ISA 08 §8.3/8.4):
//   group0: count=1 | lds_addr | global_addr | type=2
//   group1: data_size=4B, tensor_dim0 = tile_dim0 = ndw, dim0 stride = ndw
// VADDR2/VADDR3 omitted (NULL) -> 2-group descriptor, tensors up to 2D.
__device__ __forceinline__ void tdm_load_1d(const void* g, unsigned lds_addr, unsigned ndw) {
    unsigned long long ga = (unsigned long long)(size_t)g;
    v4u g0;
    g0[0] = 1u;                                   // count=1, user D#
    g0[1] = lds_addr;                             // LDS byte address
    g0[2] = (unsigned)ga;                         // global_addr[31:0]
    g0[3] = (unsigned)(ga >> 32) | 0x80000000u;   // global_addr[56:32] | type=2
    v8u g1;
    g1[0] = 2u << 16;                             // data_size = 4 bytes
    g1[1] = (ndw & 0xFFFFu) << 16;                // tensor_dim0[15:0]  (bits 63:48)
    g1[2] = (ndw >> 16) | (1u << 16);             // tensor_dim0[31:16] | tensor_dim1=1
    g1[3] = (ndw & 0xFFFFu) << 16;                // tile_dim0 (bits 127:112)
    g1[4] = 0u;                                   // tile_dim1/2 unused
    g1[5] = ndw;                                  // tensor_dim0_stride[31:0]
    g1[6] = 0u;                                   // stride hi / dim1 stride
    g1[7] = 0u;
    asm volatile("tensor_load_to_lds %0, %1" :: "s"(g0), "s"(g1) : "memory");
}

// --- Pack conv weights into per-lane A-fragment layout (contiguous k order,
// matching B, not the ISA table): lane = khalf*16 + m; element e -> K = e + 16*khalf.
// Global K order: k = (kh*3+kw)*NC + c  (chunk kc covers k in [kc*32, kc*32+32)).
__global__ void prep_weights_k(const float* __restrict__ w,
                               __bf16* __restrict__ wph, __bf16* __restrict__ wpl) {
    int idx = blockIdx.x * 256 + threadIdx.x;
    if (idx >= WPK) return;
    int e     = idx & 15;
    int lane  = (idx >> 4) & 31;
    int kc    = idx >> 9;
    int khalf = lane >> 4;
    int m     = lane & 15;                      // output row = region s
    int K     = e + 16 * khalf;
    int k     = kc * 32 + K;
    int khkw  = k / NC;
    int c     = k - khkw * NC;
    int kh    = khkw / 3, kw = khkw - kh * 3;
    float v = 0.0f;
    if (m < NS) v = w[((m * NC + c) * 3 + kh) * 3 + kw];
    __bf16 hi = bhi(v);
    wph[idx] = hi;
    wpl[idx] = blo(v, hi);
}

__global__ void zero_mask_k(__bf16* __restrict__ mh, __bf16* __restrict__ ml) {
    int idx = blockIdx.x * 256 + threadIdx.x;
    if (idx < NB * NS * MPAD) {
        mh[idx] = (__bf16)0.0f;
        ml[idx] = (__bf16)0.0f;
    }
}

// --- Conv as implicit GEMM: D(16x16) tile = 8 regions x 16 pixels.
// 7 waves per block (one image), LDS-staged A fragments per 3x3 tap.
__global__ __launch_bounds__(224)
void conv_wmma_k(const float* __restrict__ x, const float* __restrict__ bias,
                 const __bf16* __restrict__ wph, const __bf16* __restrict__ wpl,
                 float* __restrict__ mask_out,
                 __bf16* __restrict__ mh, __bf16* __restrict__ ml) {
    __shared__ alignas(32) __bf16 sAh[GRP];   // 24.6 kB
    __shared__ alignas(32) __bf16 sAl[GRP];   // 24.6 kB
    const int tid   = threadIdx.x;
    const int b     = blockIdx.x / 7;
    const int tq    = blockIdx.x - b * 7;
    const int wv    = tid >> 5;
    const int lane  = tid & 31;
    const int tile  = tq * 7 + wv;            // 0..48
    const int ln    = lane & 15;              // B-matrix column = pixel within tile
    const int khalf = lane >> 4;              // K half (B lanes 0-15: K 0..15, 16-31: K 16..31)
    const int pix   = tile * 16 + ln;
    const int h     = pix / 28, w = pix - h * 28;
    const float* xb = x + (size_t)b * NC * NHW;
    const unsigned ldsh = (unsigned)(size_t)(void*)sAh;
    const unsigned ldsl = (unsigned)(size_t)(void*)sAl;

    v8f acc = {};
    #pragma unroll 1
    for (int khkw = 0; khkw < 9; ++khkw) {
        // Stage this tap group's packed A fragments (hi+lo, 48 kB) into LDS:
        // hi via one TDM descriptor, lo via per-lane async copies (both DMA
        // engines overlap; waves only wait their own counters before barrier).
        __syncthreads();                       // previous group's LDS reads done
        {
            const __bf16* gh = wph + (size_t)khkw * GRP;
            const __bf16* gl = wpl + (size_t)khkw * GRP;
            if (wv == 0)
                tdm_load_1d(gh, ldsh, GRP / 2);          // 6144 dwords, 1 instr
            for (int i = tid; i < GRP / 8; i += 224)     // 1536 x 16B
                async_copy16(gl + (size_t)i * 8, ldsl + (unsigned)i * 16u);
            wait_asynccnt0();                  // own ASYNCcnt drained
            if (wv == 0)
                __builtin_amdgcn_s_wait_tensorcnt(0);    // TDM done
        }
        __syncthreads();                       // LDS visible to all waves

        const int kh = khkw / 3, kw = khkw - kh * 3;
        const int hh = h + kh - 1, ww = w + kw - 1;
        const bool valid = (hh >= 0) && (hh < 28) && (ww >= 0) && (ww < 28);
        const int spix = hh * 28 + ww;
        #pragma unroll 2
        for (int j = 0; j < 24; ++j) {
            const int apo = j * 512 + lane * 16;
            v16bf ah = *(const v16bf*)(sAh + apo);     // 2x ds_load_b128
            v16bf al = *(const v16bf*)(sAl + apo);
            // B fragment: K = e + 16*khalf -> channel c = j*32 + K, shifted pixel.
            const int cbase = j * 32 + 16 * khalf;
            const float* xp = xb + (size_t)cbase * NHW + spix;
            v16bf bhv, blv;
            #pragma unroll
            for (int e = 0; e < 16; ++e) {
                float v = valid ? xp[(size_t)e * NHW] : 0.0f;
                __bf16 hv = bhi(v);
                bhv[e] = hv;
                blv[e] = blo(v, hv);
            }
            acc = __builtin_amdgcn_wmma_f32_16x16x32_bf16(false, ah, false, bhv, (short)0, acc, false, false);
            acc = __builtin_amdgcn_wmma_f32_16x16x32_bf16(false, ah, false, blv, (short)0, acc, false, false);
            acc = __builtin_amdgcn_wmma_f32_16x16x32_bf16(false, al, false, bhv, (short)0, acc, false, false);
        }
    }
    // D layout: lanes 0-15 hold M = r (rows 0..7 = valid regions).
    if (khalf == 0) {
        #pragma unroll
        for (int r = 0; r < NS; ++r) {
            const float v = acc[r] + bias[r];
            const size_t mi = (size_t)b * NS + r;
            mask_out[mi * NHW + pix] = v;
            __bf16 hv = bhi(v);
            mh[mi * MPAD + pix] = hv;
            ml[mi * MPAD + pix] = blo(v, hv);
        }
    }
}

// --- Pooling einsum as per-batch GEMM: D(16x16) = mask(16xK) * x(Kx16),
// M = regions (8 valid), N = 16 channels, K = 784 padded to 800.
__global__ __launch_bounds__(32)
void pool_wmma_k(const float* __restrict__ x,
                 const __bf16* __restrict__ mh, const __bf16* __restrict__ ml,
                 float* __restrict__ learned) {
    const int b     = blockIdx.x / 48;
    const int ct    = blockIdx.x - b * 48;
    const int lane  = threadIdx.x;
    const int ln    = lane & 15;
    const int khalf = lane >> 4;
    const int c     = ct * 16 + ln;              // B-matrix column
    const int srow  = ln & 7;                    // A row (rows 8-15 duplicate, unused)
    const __bf16* mrh = mh + ((size_t)b * NS + srow) * MPAD;
    const __bf16* mrl = ml + ((size_t)b * NS + srow) * MPAD;
    const float*  xrow = x + ((size_t)b * NC + c) * NHW;

    v8f acc = {};
    #pragma unroll 1
    for (int pc = 0; pc < 25; ++pc) {
        // A fragment: elements 0..15 -> K = 0..15 + 16*khalf.
        const int abase = pc * 32 + 16 * khalf;
        v8bf a0h = *(const v8bf*)(mrh + abase);
        v8bf a1h = *(const v8bf*)(mrh + abase + 8);
        v8bf a0l = *(const v8bf*)(mrl + abase);
        v8bf a1l = *(const v8bf*)(mrl + abase + 8);
        v16bf ah = __builtin_shufflevector(a0h, a1h, 0,1,2,3,4,5,6,7,8,9,10,11,12,13,14,15);
        v16bf al = __builtin_shufflevector(a0l, a1l, 0,1,2,3,4,5,6,7,8,9,10,11,12,13,14,15);
        // B fragment: K = e + 16*khalf -> pix = pc*32 + K, 16 contiguous f32.
        v16bf bhv, blv;
        if ((pc < 24) || (khalf == 0)) {
            const v4f* xp = (const v4f*)(xrow + pc * 32 + 16 * khalf);
            __builtin_prefetch(xrow + pc * 32 + 128, 0, 0);   // stream-ahead hint
            #pragma unroll
            for (int q = 0; q < 4; ++q) {
                v4f f = xp[q];
                #pragma unroll
                for (int t = 0; t < 4; ++t) {
                    __bf16 hv = bhi(f[t]);
                    bhv[q * 4 + t] = hv;
                    blv[q * 4 + t] = blo(f[t], hv);
                }
            }
        } else {
            #pragma unroll
            for (int e = 0; e < 16; ++e) { bhv[e] = (__bf16)0.0f; blv[e] = (__bf16)0.0f; }
        }
        acc = __builtin_amdgcn_wmma_f32_16x16x32_bf16(false, ah, false, bhv, (short)0, acc, false, false);
        acc = __builtin_amdgcn_wmma_f32_16x16x32_bf16(false, ah, false, blv, (short)0, acc, false, false);
        acc = __builtin_amdgcn_wmma_f32_16x16x32_bf16(false, al, false, bhv, (short)0, acc, false, false);
    }
    if (khalf == 0) {
        #pragma unroll
        for (int r = 0; r < NS; ++r) {
            learned[((size_t)b * NC + c) * NS + r] = acc[r] * (1.0f / 784.0f);
        }
    }
}

extern "C" void kernel_launch(void* const* d_in, const int* in_sizes, int n_in,
                              void* d_out, int out_size, void* d_ws, size_t ws_size,
                              hipStream_t stream) {
    const float* x      = (const float*)d_in[0];   // [64,768,28,28]
    const float* conv_w = (const float*)d_in[1];   // [8,768,3,3]
    const float* conv_b = (const float*)d_in[2];   // [8]
    float* out      = (float*)d_out;
    float* learned  = out;                 // [64,768,8]  = 393216 f32
    float* mask_out = out + NB * NC * NS;  // [64,8,28,28]= 401408 f32

    __bf16* wsb = (__bf16*)d_ws;           // ~2.1 MB used
    __bf16* wph = wsb;
    __bf16* wpl = wph + WPK;
    __bf16* mh  = wpl + WPK;
    __bf16* ml  = mh + (size_t)NB * NS * MPAD;

    prep_weights_k<<<(WPK + 255) / 256, 256, 0, stream>>>(conv_w, wph, wpl);
    zero_mask_k<<<(NB * NS * MPAD + 255) / 256, 256, 0, stream>>>(mh, ml);
    conv_wmma_k<<<NB * 7, 224, 0, stream>>>(x, conv_b, wph, wpl, mask_out, mh, ml);
    pool_wmma_k<<<NB * 48, 32, 0, stream>>>(x, mh, ml, learned);
}
